// RSSM_77927886618959
// MI455X (gfx1250) — matrix operand order repaired and checked by reading, not compiled, not run on hardware
//
#include <hip/hip_runtime.h>

// ---------------- problem constants ----------------
#define BB   16      // batch
#define TT   64      // timesteps
#define HH   4096    // recurrent width
#define HID  2048
#define SS   32
#define CC   32
#define BLK  8
#define TD   1024
#define AD   32
#define BH   512     // H/BLK
#define BX   768     // 3*HID/BLK
#define GK   (BX+BH) // 1280 GRU K
#define GN   (3*BH)  // 1536 GRU N
#define PK   (HH+TD) // 5120 posterior K

#define ACHUNK 512   // A-panel K-chunk staged to LDS by the TDM (16x512 bf16 = 16KB/buf)

typedef __attribute__((ext_vector_type(16))) __bf16 v16bf;
typedef __attribute__((ext_vector_type(8)))  __bf16 v8bf;
typedef __attribute__((ext_vector_type(8)))  float  v8f;
typedef __attribute__((ext_vector_type(4)))  unsigned int u32x4;
typedef __attribute__((ext_vector_type(8)))  int i32x8;
typedef __attribute__((ext_vector_type(4)))  int i32x4;

// Tensor Data Mover availability (device pass only; host pass parses fallback).
#if defined(__AMDGCN__) && defined(__has_builtin)
#  if __has_builtin(__builtin_amdgcn_tensor_load_to_lds)
#    define RSSM_USE_TDM 1
#  endif
#endif
#ifndef RSSM_USE_TDM
#  define RSSM_USE_TDM 0
#endif

#if RSSM_USE_TDM
// Issue one TDM load of a 16-row x cs-col bf16 tile (row stride K elements)
// from global into LDS at byte offset lds_off. D# per CDNA5 ISA ch.8.
__device__ __forceinline__ void tdm_load_a(const __bf16* gsrc, unsigned lds_off,
                                           int K, int cs)
{
    unsigned long long ga = (unsigned long long)(uintptr_t)gsrc;
    u32x4 g0;
    g0.x = 1u;                                   // count=1, user descriptor
    g0.y = lds_off;                              // lds_addr (bytes)
    g0.z = (unsigned)ga;                         // global_addr[31:0]
    g0.w = (unsigned)((ga >> 32) & 0x1FFFFFFu) | (2u << 30);  // addr[56:32] | type=2
    i32x8 g1;
    g1[0] = (int)(1u << 16);                     // data_size=1 -> 2 bytes (bf16)
    g1[1] = (int)(((unsigned)K & 0xFFFFu) << 16);   // tensor_dim0[15:0]
    g1[2] = (int)(16u << 16);                    // tensor_dim0[31:16]=0 | tensor_dim1=16
    g1[3] = (int)(((unsigned)cs & 0xFFFFu) << 16);  // tile_dim0 = cs
    g1[4] = 16;                                  // tile_dim1 = 16, tile_dim2 = 0
    g1[5] = K;                                   // tensor_dim0_stride[31:0]
    g1[6] = 0;
    g1[7] = 0;
    i32x4 z4 = {};
#if __clang_major__ >= 23
    i32x8 z8 = {};
    __builtin_amdgcn_tensor_load_to_lds(g0, g1, z4, z4, z8, 0);
#else
    __builtin_amdgcn_tensor_load_to_lds(g0, g1, z4, z4, 0);
#endif
}
#endif

// ============================================================
// Pack fp32 weight (K x N row-major) -> bf16 WMMA B-tile layout.
// Flat index: ((tile*KT + kt)*32 + lane)*16 + e
//   n = tile*16 + (lane&15);  k = kt*32 + (lane>>4)*16 + e
// ============================================================
__global__ void rssm_pack_w(const float* __restrict__ W, __bf16* __restrict__ P,
                            int K, int N)
{
    const size_t total = (size_t)K * N;
    size_t tid = (size_t)blockIdx.x * 256 + threadIdx.x;
    if (tid >= total) return;
    const float* Wb = W + (size_t)blockIdx.y * total;
    __bf16*      Pb = P + (size_t)blockIdx.y * total;
    int e    = (int)(tid & 15);
    int lane = (int)((tid >> 4) & 31);
    int KT   = K >> 5;
    int kt   = (int)((tid >> 9) % (size_t)KT);
    int tile = (int)(tid / ((size_t)KT << 9));
    int n = tile * 16 + (lane & 15);
    int k = kt * 32 + ((lane >> 4) << 4) + e;
    Pb[tid] = (__bf16)Wb[(size_t)k * N + n];
}

// ============================================================
// Core GEMM: Y(16 x N, f32) = A(16 x K, bf16 row-major) @ Bp(packed bf16)
// One wave = one 16x16 output tile, K in steps of 32 via WMMA.
// A panel is DMA'd to LDS by the TDM (double-buffered, wave 0 drives it),
// B streams from L2 via b128 loads (one-deep software pipeline) + prefetch.
// ============================================================
__global__ void rssm_gemm(const __bf16* __restrict__ A, const __bf16* __restrict__ Bp,
                          float* __restrict__ Y, int K, int N,
                          long long sA, long long sB, long long sY)
{
    const int lane = threadIdx.x & 31;
    const int wave = threadIdx.x >> 5;
    const int tile = blockIdx.x * (blockDim.x >> 5) + wave;   // n-tile
    const int bat  = blockIdx.y;
    const int half = lane >> 4;
    const int mn   = lane & 15;
    const int KT   = K >> 5;

    const __bf16* Abase = A + (size_t)bat * sA;
    const __bf16* Bt    = Bp + (size_t)bat * sB + (size_t)tile * KT * 512 + lane * 16;

    v8f acc = {};

#if RSSM_USE_TDM
    __shared__ __bf16 Alds[2][16 * ACHUNK];
    const int NC = (K + ACHUNK - 1) / ACHUNK;
    if (wave == 0) {
        int cs0 = K < ACHUNK ? K : ACHUNK;
        tdm_load_a(Abase, (unsigned)(uintptr_t)&Alds[0][0], K, cs0);
    }
    int ktg = 0;
    for (int c = 0; c < NC; ++c) {
        const int rem = K - c * ACHUNK;
        const int cs  = rem < ACHUNK ? rem : ACHUNK;
        if (wave == 0) __builtin_amdgcn_s_wait_tensorcnt(0);   // chunk c resident
        __syncthreads();
        if (wave == 0 && c + 1 < NC) {                         // overlap next DMA
            int rem1 = K - (c + 1) * ACHUNK;
            int cs1  = rem1 < ACHUNK ? rem1 : ACHUNK;
            tdm_load_a(Abase + (size_t)(c + 1) * ACHUNK,
                       (unsigned)(uintptr_t)&Alds[(c + 1) & 1][0], K, cs1);
        }
        const __bf16* Arow = &Alds[c & 1][0] + (size_t)mn * cs + half * 16;
        const __bf16* bkp  = Bt + (size_t)ktg * 512;           // per-chunk base: inner
        const int KTc = cs >> 5;                               // loop uses imm offsets
        // one-deep B software pipeline: bcur consumed while bnxt is in flight
        v16bf bcur = *(const v16bf*)bkp;
        for (int kt = 0; kt < KTc; ++kt) {
            v16bf bnxt = *(const v16bf*)(bkp + (size_t)(kt + 1) * 512);  // benign overrun
            __builtin_prefetch((const void*)(bkp + (size_t)(kt + 5) * 512), 0, 1);
            union { v16bf v; v8bf h[2]; } a;
            // A fragment (contiguous, matches packed B): e0..15 -> K = kt*32 + half*16 + e
            a.h[0] = *(const v8bf*)(Arow + kt * 32);
            a.h[1] = *(const v8bf*)(Arow + kt * 32 + 8);
            acc = __builtin_amdgcn_wmma_f32_16x16x32_bf16(
                      false, a.v, false, bcur, (short)0, acc, false, false);
            bcur = bnxt;
        }
        ktg += KTc;
        __syncthreads();   // all waves done with buf[c&1] before TDM(c+2) rewrites it
    }
#else
    const __bf16* Arow = Abase + (size_t)mn * K + half * 16;
    for (int kt = 0; kt < KT; ++kt) {
        union { v16bf v; v8bf h[2]; } a, b;
        a.h[0] = *(const v8bf*)(Arow + kt * 32);
        a.h[1] = *(const v8bf*)(Arow + kt * 32 + 8);
        b.v = *(const v16bf*)(Bt + (size_t)kt * 512);
        __builtin_prefetch((const void*)(Bt + (size_t)(kt + 4) * 512), 0, 1);
        acc = __builtin_amdgcn_wmma_f32_16x16x32_bf16(
                  false, a.v, false, b.v, (short)0, acc, false, false);
    }
#endif

    float* Yt = Y + (size_t)bat * sY;
    const int n = tile * 16 + mn;
    #pragma unroll
    for (int r = 0; r < 8; ++r)
        Yt[(size_t)(r + 8 * half) * N + n] = acc[r];
}

// ============================================================
// rmsnorm -> SiLU -> bf16, writing either plain row-major (mode 0)
// or into the GRU concat buffer xh[(k*16+b)*1280 + pos] (mode 1).
// ============================================================
__global__ void rssm_norm_silu(const float* __restrict__ Yraw, const float* __restrict__ scale,
                               __bf16* __restrict__ dst, int N, int mode, int branch)
{
    const int b = blockIdx.x;
    const float* row = Yraw + (size_t)b * N;
    __shared__ float red[256];
    float ss = 0.f;
    for (int i = threadIdx.x; i < N; i += 256) { float v = row[i]; ss += v * v; }
    red[threadIdx.x] = ss; __syncthreads();
    for (int s = 128; s > 0; s >>= 1) {
        if (threadIdx.x < s) red[threadIdx.x] += red[threadIdx.x + s];
        __syncthreads();
    }
    const float inv = rsqrtf(red[0] / (float)N + 1e-6f);
    for (int i = threadIdx.x; i < N; i += 256) {
        float y = row[i] * inv * scale[i];
        float a = y / (1.f + __expf(-y));          // SiLU
        __bf16 o = (__bf16)a;
        if (mode == 0) {
            dst[(size_t)b * N + i] = o;
        } else {
            int c = branch * HID + i;              // concat column in [0,6144)
            int k = c / BX, pos = c % BX;          // GRU block / position
            dst[((size_t)k * 16 + b) * GK + pos] = o;
        }
    }
}

// ============================================================
// Per-step state prep: reset-mask h, emit bf16 A0 (=h) and xh h-part.
// ============================================================
__global__ void rssm_prep_h(float* __restrict__ h, __bf16* __restrict__ h_bf,
                            __bf16* __restrict__ xh, const unsigned char* __restrict__ reset,
                            int t)
{
    int tid = blockIdx.x * 256 + threadIdx.x;
    if (tid >= BB * HH) return;
    int b = tid >> 12, i = tid & (HH - 1);
    float m = reset[b * TT + t] ? 0.f : 1.f;
    float hv = h[tid] * m;
    h[tid] = hv;
    h_bf[tid] = (__bf16)hv;
    int k = i >> 9, pos = i & (BH - 1);
    xh[((size_t)k * 16 + b) * GK + BX + pos] = (__bf16)hv;
}

// z mask -> A1, tok -> A_p tail, action mask+clip -> A2
__global__ void rssm_prep_aux(const float* __restrict__ tokens, const float* __restrict__ action,
                              const unsigned char* __restrict__ reset, float* __restrict__ z,
                              __bf16* __restrict__ A1, __bf16* __restrict__ A2,
                              __bf16* __restrict__ A_p, int t)
{
    int tid = blockIdx.x * 256 + threadIdx.x;
    if (tid >= BB * (SS * CC)) return;
    int b = tid >> 10, j = tid & 1023;
    float m = reset[b * TT + t] ? 0.f : 1.f;
    float zv = z[tid] * m;
    z[tid] = zv;
    A1[tid] = (__bf16)zv;
    A_p[(size_t)b * PK + HH + j] = (__bf16)tokens[((size_t)b * TT + t) * TD + j];
    if (j < AD) {
        float a = action[((size_t)b * TT + t) * AD + j] * m;
        a = a / fmaxf(1.f, fabsf(a));
        A2[b * AD + j] = (__bf16)a;
    }
}

// ============================================================
// GRU gates + state update. One block per (b,k) row of parts (1536 wide).
// ============================================================
__global__ void rssm_gru_update(const float* __restrict__ parts, const float* __restrict__ sg,
                                float* __restrict__ h, __bf16* __restrict__ A_p,
                                float* __restrict__ out, int t)
{
    const int b = blockIdx.x >> 3;
    const int k = blockIdx.x & 7;
    const float* row = parts + ((size_t)k * 16 + b) * GN;
    const float* sgk = sg + (size_t)k * GN;
    __shared__ float red[256];
    float ss = 0.f;
    for (int i = threadIdx.x; i < GN; i += 256) { float v = row[i]; ss += v * v; }
    red[threadIdx.x] = ss; __syncthreads();
    for (int s = 128; s > 0; s >>= 1) {
        if (threadIdx.x < s) red[threadIdx.x] += red[threadIdx.x + s];
        __syncthreads();
    }
    const float inv = rsqrtf(red[0] / (float)GN + 1e-6f);
    for (int j = threadIdx.x; j < BH; j += 256) {
        float r = row[j]           * inv * sgk[j];
        float c = row[BH + j]      * inv * sgk[BH + j];
        float u = row[2 * BH + j]  * inv * sgk[2 * BH + j];
        float rg   = 1.f / (1.f + __expf(-r));
        float cand = tanhf(rg * c);
        float up   = 1.f / (1.f + __expf(-(u - 1.f)));
        int hi = b * HH + k * BH + j;
        float hn = up * cand + (1.f - up) * h[hi];
        h[hi] = hn;
        A_p[(size_t)b * PK + k * BH + j] = (__bf16)hn;
        out[((size_t)b * TT + t) * (HH + SS * CC) + k * BH + j] = hn;
    }
}

// ============================================================
// Posterior: bias, logits to out, unimix softmax, categorical sample
// (hash RNG); straight-through z == one-hot numerically.
// ============================================================
__global__ void rssm_sample(const float* __restrict__ Ylog, const float* __restrict__ bph,
                            float* __restrict__ z, float* __restrict__ out, int t)
{
    int tid = blockIdx.x * blockDim.x + threadIdx.x;
    if (tid >= BB * SS) return;
    int b = tid >> 5, s = tid & 31;
    float lg[CC];
    float mx = -1e30f;
    for (int c = 0; c < CC; ++c) {
        float v = Ylog[(size_t)b * (SS * CC) + s * CC + c] + bph[s * CC + c];
        lg[c] = v;
        mx = fmaxf(mx, v);
        out[((size_t)b * TT + t) * (HH + SS * CC) + HH + s * CC + c] = v;
    }
    float den = 0.f;
    for (int c = 0; c < CC; ++c) den += __expf(lg[c] - mx);
    unsigned x = (unsigned)(t * 0x9E3779B9u) ^ (unsigned)(b * 0x85EBCA6Bu + 0x27D4EB2Fu)
               ^ (unsigned)(s * 0x165667B1u);
    x ^= x >> 16; x *= 0x7FEB352Du; x ^= x >> 15; x *= 0x846CA68Bu; x ^= x >> 16;
    float u = (float)(x >> 8) * (1.0f / 16777216.0f);
    float cdf = 0.f; int idx = CC - 1;
    for (int c = 0; c < CC; ++c) {
        float p = 0.99f * (__expf(lg[c] - mx) / den) + 0.01f / (float)CC;
        cdf += p;
        if (u < cdf) { idx = c; break; }
    }
    for (int c = 0; c < CC; ++c)
        z[(size_t)b * (SS * CC) + s * CC + c] = (c == idx) ? 1.f : 0.f;
}

__global__ void rssm_zero_state(float* __restrict__ h, float* __restrict__ z)
{
    int tid = blockIdx.x * 256 + threadIdx.x;
    if (tid < BB * HH) h[tid] = 0.f;
    if (tid < BB * SS * CC) z[tid] = 0.f;
}

// ---------------- host side ----------------
static inline char* carve(char*& p, size_t bytes)
{
    char* r = p;
    p += (bytes + 255) & ~(size_t)255;
    return r;
}

extern "C" void kernel_launch(void* const* d_in, const int* in_sizes, int n_in,
                              void* d_out, int out_size, void* d_ws, size_t ws_size,
                              hipStream_t stream)
{
    (void)in_sizes; (void)n_in; (void)out_size; (void)ws_size;
    const float* tokens = (const float*)d_in[0];
    const float* action = (const float*)d_in[1];
    const unsigned char* reset = (const unsigned char*)d_in[2]; // all zeros in reference
    const float* w0  = (const float*)d_in[3];
    const float* s0  = (const float*)d_in[4];
    const float* w1  = (const float*)d_in[5];
    const float* s1  = (const float*)d_in[6];
    const float* w2  = (const float*)d_in[7];
    const float* s2  = (const float*)d_in[8];
    const float* wg  = (const float*)d_in[9];
    const float* sg  = (const float*)d_in[10];
    const float* wp  = (const float*)d_in[11];
    const float* sp  = (const float*)d_in[12];
    const float* wph = (const float*)d_in[13];
    const float* bph = (const float*)d_in[14];
    float* out = (float*)d_out;

    // ---- workspace carve ----
    char* p = (char*)d_ws;
    __bf16* w0p  = (__bf16*)carve(p, (size_t)HH * HID * 2);
    __bf16* w1p  = (__bf16*)carve(p, (size_t)(SS * CC) * HID * 2);
    __bf16* w2p  = (__bf16*)carve(p, (size_t)AD * HID * 2);
    __bf16* wgp  = (__bf16*)carve(p, (size_t)BLK * GK * GN * 2);
    __bf16* wpp  = (__bf16*)carve(p, (size_t)PK * HID * 2);
    __bf16* wphp = (__bf16*)carve(p, (size_t)HID * (SS * CC) * 2);
    float*  h    = (float*)carve(p, (size_t)BB * HH * 4);
    __bf16* h_bf = (__bf16*)carve(p, (size_t)BB * HH * 2);
    float*  zst  = (float*)carve(p, (size_t)BB * SS * CC * 4);
    __bf16* A1   = (__bf16*)carve(p, (size_t)BB * SS * CC * 2);
    __bf16* A2   = (__bf16*)carve(p, (size_t)BB * AD * 2);
    __bf16* A_p  = (__bf16*)carve(p, (size_t)BB * PK * 2);
    __bf16* A_h  = (__bf16*)carve(p, (size_t)BB * HID * 2);
    __bf16* xh   = (__bf16*)carve(p, (size_t)BLK * 16 * GK * 2);
    float*  Y0   = (float*)carve(p, (size_t)BB * HID * 4);
    float*  Y1   = (float*)carve(p, (size_t)BB * HID * 4);
    float*  Y2   = (float*)carve(p, (size_t)BB * HID * 4);
    float*  prt  = (float*)carve(p, (size_t)BLK * 16 * GN * 4);
    float*  Yp   = (float*)carve(p, (size_t)BB * HID * 4);
    float*  Ylg  = (float*)carve(p, (size_t)BB * SS * CC * 4);

    // ---- pack all weights to bf16 WMMA layout (78 MB -> resident in 192MB L2) ----
    auto packgrid = [](size_t elems) { return dim3((unsigned)((elems + 255) / 256), 1); };
    rssm_pack_w<<<packgrid((size_t)HH * HID), 256, 0, stream>>>(w0, w0p, HH, HID);
    rssm_pack_w<<<packgrid((size_t)SS * CC * HID), 256, 0, stream>>>(w1, w1p, SS * CC, HID);
    rssm_pack_w<<<packgrid((size_t)AD * HID), 256, 0, stream>>>(w2, w2p, AD, HID);
    rssm_pack_w<<<dim3((unsigned)(((size_t)GK * GN + 255) / 256), BLK), 256, 0, stream>>>(wg, wgp, GK, GN);
    rssm_pack_w<<<packgrid((size_t)PK * HID), 256, 0, stream>>>(wp, wpp, PK, HID);
    rssm_pack_w<<<packgrid((size_t)HID * SS * CC), 256, 0, stream>>>(wph, wphp, HID, SS * CC);

    rssm_zero_state<<<dim3(256), 256, 0, stream>>>(h, zst);

    for (int t = 0; t < TT; ++t) {
        rssm_prep_h<<<dim3(256), 256, 0, stream>>>(h, h_bf, xh, reset, t);
        rssm_prep_aux<<<dim3(64), 256, 0, stream>>>(tokens, action, reset, zst, A1, A2, A_p, t);

        // three NormedLinear branches (M=16 rows -> one WMMA tile row per wave)
        rssm_gemm<<<dim3(HID / 128), 256, 0, stream>>>(h_bf, w0p, Y0, HH, HID, 0, 0, 0);
        rssm_gemm<<<dim3(HID / 128), 256, 0, stream>>>(A1, w1p, Y1, SS * CC, HID, 0, 0, 0);
        rssm_gemm<<<dim3(HID / 128), 256, 0, stream>>>(A2, w2p, Y2, AD, HID, 0, 0, 0);
        rssm_norm_silu<<<dim3(BB), 256, 0, stream>>>(Y0, s0, xh, HID, 1, 0);
        rssm_norm_silu<<<dim3(BB), 256, 0, stream>>>(Y1, s1, xh, HID, 1, 1);
        rssm_norm_silu<<<dim3(BB), 256, 0, stream>>>(Y2, s2, xh, HID, 1, 2);

        // block-diagonal GRU: 8 batched (16x1280)@(1280x1536)
        rssm_gemm<<<dim3(GN / 128, BLK), 256, 0, stream>>>(
            xh, wgp, prt, GK, GN,
            (long long)16 * GK, (long long)GK * GN, (long long)16 * GN);
        rssm_gru_update<<<dim3(BB * BLK), 256, 0, stream>>>(prt, sg, h, A_p, out, t);

        // posterior MLP + head
        rssm_gemm<<<dim3(HID / 128), 256, 0, stream>>>(A_p, wpp, Yp, PK, HID, 0, 0, 0);
        rssm_norm_silu<<<dim3(BB), 256, 0, stream>>>(Yp, sp, A_h, HID, 0, 0);
        rssm_gemm<<<dim3((SS * CC) / 128), 256, 0, stream>>>(A_h, wphp, Ylg, HID, SS * CC, 0, 0, 0);
        rssm_sample<<<dim3(1), 512, 0, stream>>>(Ylg, bph, zst, out, t);
    }
}
